// ODENet_66228395705120
// MI455X (gfx1250) — compile-verified, matrix operand order checked by reading, not run on hardware
//
#include <hip/hip_runtime.h>

#define SEQ_LEN 16384
#define BATCH   128
#define HIDDEN  128

typedef __attribute__((ext_vector_type(16))) _Float16 v16h;
typedef __attribute__((ext_vector_type(8)))  _Float16 v8h;
typedef __attribute__((ext_vector_type(8)))  float    v8f;
typedef __attribute__((ext_vector_type(4)))  float    v4f;

union V16 { v16h v; v8h h[2]; _Float16 e[16]; };

__device__ __forceinline__ float fast_tanh(float x) {
#if __has_builtin(__builtin_amdgcn_tanhf)
    return __builtin_amdgcn_tanhf(x);          // v_tanh_f32 (CDNA5 TRANS op)
#else
    x = fminf(fmaxf(x, -15.0f), 15.0f);
    float e = __expf(x + x);
    return fmaf(-2.0f, __builtin_amdgcn_rcpf(e + 1.0f), 1.0f);
#endif
}

__device__ __forceinline__ void sched_fence() {
#if __has_builtin(__builtin_amdgcn_sched_barrier)
    __builtin_amdgcn_sched_barrier(0);         // no instruction may cross
#else
    asm volatile("" ::: "memory");
#endif
}

// One workgroup per 16-row batch slice (8 WGPs busy). 8 waves.
// Layer 2 computed TRANSPOSED: h2^T = W2^T @ h1^T. Wave w owns M-tile
// n in [w*16, w*16+16); W2^T A-fragments persist in registers for the whole
// 16384-step scan; h1^T is the B operand (column m of h1^T = row m of h1,
// contiguous in LDS -> two ds_load_b128 per K-tile). The C fragment gives
// each lane 8 h2 values of ONE batch row m across n, so the layer-3 dot
// product with W3 is 8 in-lane FMAs + one shfl_xor(16) + one ds_store_b32.
// State y[m] lives in a register (replicated across threads sharing row m).
__global__ __launch_bounds__(256, 1)
void odenet_scan_kernel(const float* __restrict__ x,
                        const float* __restrict__ W1,
                        const float* __restrict__ b1,
                        const float* __restrict__ W2,
                        const float* __restrict__ b2,
                        const float* __restrict__ W3,
                        const float* __restrict__ b3,
                        float* __restrict__ out)
{
    __shared__ __align__(16) _Float16 h1s[16 * 136];  // 16 rows, stride 136 halves
    __shared__ __align__(16) float    vpart[16 * 8];  // [m][wave]

    const int t    = threadIdx.x;
    const int lane = t & 31;
    const int wid  = t >> 5;
    const int b0   = blockIdx.x * 16;

    // ---- layer-1 role: thread t computes h1[m][jbase..jbase+7] ----
    const int m     = t & 15;                  // == lane & 15
    const int jbase = (t >> 4) * 8;
    float w1a[8], w1b[8], bb1[8];
    #pragma unroll
    for (int i = 0; i < 8; ++i) {
        w1a[i] = W1[jbase + i];           // weight on y
        w1b[i] = W1[HIDDEN + jbase + i];  // weight on x
        bb1[i] = b1[jbase + i];
    }

    // ---- GEMM role (transposed): lane owns A row n = wid*16 + (lane&15) ----
    const int   n   = wid * 16 + (lane & 15);
    const float b3v = b3[0];
    const int   kb  = (lane < 16) ? 0 : 8;    // 16-bit fragment K-base per half-wave

    // Persistent W2^T A-fragments: A'[n][j] = W2[j][n].
    // half h -> j = kt*32 + kb + (h<8 ? h : h+8)
    V16 afrag[4];
    #pragma unroll
    for (int kt = 0; kt < 4; ++kt) {
        #pragma unroll
        for (int h = 0; h < 16; ++h) {
            int j = kt * 32 + kb + ((h < 8) ? h : (h + 8));
            afrag[kt].e[h] = (_Float16)W2[j * HIDDEN + n];
        }
    }

    // Per-lane n values of the C fragment: n_g = wid*16 + kb + g  (M = vgpr idx)
    float b2v[8], w3v[8];
    #pragma unroll
    for (int g = 0; g < 8; ++g) {
        int ng = wid * 16 + kb + g;
        b2v[g] = b2[ng];
        w3v[g] = W3[ng];
    }

    float yreg = 0.0f;   // ODE state for row m (replicated, deterministic)
    __syncthreads();

    for (int s = 0; s < SEQ_LEN; ++s) {
        // ---------- layer 1: h1 = tanh(y*W1[0] + x*W1[1] + b1) -> f16 LDS ----------
        float xv = x[s * BATCH + b0 + m];
        if (s + 1 < SEQ_LEN)
            __builtin_prefetch(&x[(s + 1) * BATCH + b0 + m], 0, 0); // global_prefetch_b8
        v8h h1v;
        #pragma unroll
        for (int i = 0; i < 8; ++i)
            h1v[i] = (_Float16)fast_tanh(fmaf(yreg, w1a[i], fmaf(xv, w1b[i], bb1[i])));
        *(v8h*)&h1s[m * 136 + jbase] = h1v;        // one ds_store_b128
        __syncthreads();                            // barrier 1

        // ---------- layer 2: issue ALL fragment loads, then 4 chained WMMAs ----------
        V16 bm0, bm1, bm2, bm3;
        {
            const _Float16* p = &h1s[(lane & 15) * 136 + kb];
            bm0.h[0] = *(const v8h*)(p +   0);  bm0.h[1] = *(const v8h*)(p +  16);
            bm1.h[0] = *(const v8h*)(p +  32);  bm1.h[1] = *(const v8h*)(p +  48);
            bm2.h[0] = *(const v8h*)(p +  64);  bm2.h[1] = *(const v8h*)(p +  80);
            bm3.h[0] = *(const v8h*)(p +  96);  bm3.h[1] = *(const v8h*)(p + 112);
        }
        sched_fence();   // keep all 8 ds_load_b128 ahead of the WMMA chain
        v8f acc;
        acc = __builtin_amdgcn_wmma_f32_16x16x32_f16(
                  false, afrag[0].v, false, bm0.v, (short)0, v8f{},  false, false);
        acc = __builtin_amdgcn_wmma_f32_16x16x32_f16(
                  false, afrag[1].v, false, bm1.v, (short)0, acc, false, false);
        acc = __builtin_amdgcn_wmma_f32_16x16x32_f16(
                  false, afrag[2].v, false, bm2.v, (short)0, acc, false, false);
        acc = __builtin_amdgcn_wmma_f32_16x16x32_f16(
                  false, afrag[3].v, false, bm3.v, (short)0, acc, false, false);

        // ---------- layer 3: in-lane dot product over this lane's 8 n values ----------
        float partial = 0.0f;
        #pragma unroll
        for (int g = 0; g < 8; ++g)
            partial = fmaf(fast_tanh(acc[g] + b2v[g]), w3v[g], partial);
        partial += __shfl_xor(partial, 16, 32);     // both halves now hold the tile sum
        vpart[m * 8 + wid] = partial;               // branch-free: dup lanes write same value
        __syncthreads();                            // barrier 2

        // ---------- Euler update (every thread, for its own row m) ----------
        v4f s0 = *(const v4f*)&vpart[m * 8];
        v4f s1 = *(const v4f*)&vpart[m * 8 + 4];    // two ds_load_b128
        float v = b3v + ((s0[0] + s0[1]) + (s0[2] + s0[3]))
                      + ((s1[0] + s1[1]) + (s1[2] + s1[3]));
        yreg += v;                                   // DT = 1.0
        if (t < 16)
            out[s * BATCH + b0 + t] = yreg;          // t == m here
        // next iteration's barrier 1 orders vpart reads vs. next writes
    }
}

extern "C" void kernel_launch(void* const* d_in, const int* in_sizes, int n_in,
                              void* d_out, int out_size, void* d_ws, size_t ws_size,
                              hipStream_t stream) {
    (void)in_sizes; (void)n_in; (void)d_ws; (void)ws_size; (void)out_size;
    const float* x  = (const float*)d_in[0];
    const float* W1 = (const float*)d_in[1];
    const float* b1 = (const float*)d_in[2];
    const float* W2 = (const float*)d_in[3];
    const float* b2 = (const float*)d_in[4];
    const float* W3 = (const float*)d_in[5];
    const float* b3 = (const float*)d_in[6];
    float* out = (float*)d_out;

    odenet_scan_kernel<<<dim3(BATCH / 16), dim3(256), 0, stream>>>(
        x, W1, b1, W2, b2, W3, b3, out);
}
